// LanguageModel_2860448219402
// MI455X (gfx1250) — compile-verified, matrix operand order checked by reading, not run on hardware
//
#include <hip/hip_runtime.h>
#include <hip/hip_bf16.h>
#include <math.h>

// ---------------------------------------------------------------------------
// Model dims (fixed by the reference)
// ---------------------------------------------------------------------------
#define BB   2
#define TT   1024
#define DD   512
#define HH   8
#define HD   64
#define FFD  2048
#define LL   4
#define VV   50257
#define NTOK (BB * TT)          // 2048 rows of activations

typedef __attribute__((ext_vector_type(16))) __bf16 v16bf;
typedef __attribute__((ext_vector_type(8)))  __bf16 v8bf;
typedef __attribute__((ext_vector_type(8)))  float  v8f;

// Fragment load: ISA 16-bit layout needs per-lane K-runs [kb..kb+7] and
// [kb+16..kb+23]; with K-contiguous LDS tiles these are two aligned 16-byte
// loads (ds_load_b128) concatenated by shufflevector (no scalar packing).
__device__ __forceinline__ v16bf ld_frag(const __bf16* p) {
  v8bf lo = *(const v8bf*)(p);
  v8bf hi = *(const v8bf*)(p + 16);
  return __builtin_shufflevector(lo, hi, 0, 1, 2, 3, 4, 5, 6, 7,
                                 8, 9, 10, 11, 12, 13, 14, 15);
}

// Convert 8 floats -> one 16-byte bf16 store (ds_store_b128)
__device__ __forceinline__ void st8(__bf16* p, const float* f) {
  v8bf v;
#pragma unroll
  for (int i = 0; i < 8; ++i) v[i] = (__bf16)f[i];
  *(v8bf*)p = v;
}

// ---------------------------------------------------------------------------
// Embedding: x[bt, d] = tok_emb[idx[bt], d] + pos_emb[t, d]
// ---------------------------------------------------------------------------
__global__ __launch_bounds__(256) void embed_k(const int* __restrict__ idx,
                                               const float* __restrict__ tok,
                                               const float* __restrict__ pos,
                                               float* __restrict__ x) {
  int i = blockIdx.x * 256 + threadIdx.x;         // over NTOK*DD
  int bt = i >> 9;                                 // /512
  int d  = i & (DD - 1);
  int t  = bt & (TT - 1);
  x[i] = tok[(size_t)idx[bt] * DD + d] + pos[(size_t)t * DD + d];
}

// ---------------------------------------------------------------------------
// LayerNorm: one 256-thread block per row of 512
// ---------------------------------------------------------------------------
__global__ __launch_bounds__(256) void layernorm_k(const float* __restrict__ x,
                                                   const float* __restrict__ w,
                                                   const float* __restrict__ bp,
                                                   float* __restrict__ out) {
  __shared__ float red[256];
  int row = blockIdx.x, tid = threadIdx.x;
  const float* xr = x + (size_t)row * DD;
  float v0 = xr[tid], v1 = xr[tid + 256];
  red[tid] = v0 + v1;
  __syncthreads();
  for (int o = 128; o > 0; o >>= 1) {
    if (tid < o) red[tid] += red[tid + o];
    __syncthreads();
  }
  float mu = red[0] * (1.0f / DD);
  __syncthreads();
  float d0 = v0 - mu, d1 = v1 - mu;
  red[tid] = d0 * d0 + d1 * d1;
  __syncthreads();
  for (int o = 128; o > 0; o >>= 1) {
    if (tid < o) red[tid] += red[tid + o];
    __syncthreads();
  }
  float rstd = rsqrtf(red[0] * (1.0f / DD) + 1e-5f);
  float* orow = out + (size_t)row * DD;
  orow[tid]       = d0 * rstd * w[tid]       + bp[tid];
  orow[tid + 256] = d1 * rstd * w[tid + 256] + bp[tid + 256];
}

// ---------------------------------------------------------------------------
// bf16 WMMA GEMM: C[M,N] = act(A[M,K] @ Bm[K,N] + bias + resid)
// Block tile 256x64, 8 waves; each wave owns a 32x64 strip:
//   per 32-wide K-step: 2 A-frags x 4 B-frags -> 8 v_wmma.
// A tile K-contiguous; B tile stored TRANSPOSED (Bst[n][k]) so both fragment
// types load as 2 x ds_load_b128 per fragment.
// ---------------------------------------------------------------------------
__global__ __launch_bounds__(256) void gemm_bf16_wmma(
    const float* __restrict__ A, int lda,
    const float* __restrict__ Bm, int ldb,
    const float* __restrict__ bias,            // per-column, may be null
    const float* __restrict__ resid, int ldr,  // may be null
    float* __restrict__ C, int ldc,
    int M, int N, int K, int relu) {
  __shared__ alignas(16) __bf16 As[256][40];   // 256x32, stride 80B (16B mult)
  __shared__ alignas(16) __bf16 Bst[64][40];   // transposed 32x64 -> [n][k]

  int tid  = threadIdx.x;
  int wave = tid >> 5;
  int lane = tid & 31;
  int row0 = blockIdx.y * 256;
  int col0 = blockIdx.x * 64;

  int nlo  = lane & 15;                 // frag row (A) / col (B/C) selector
  int kb   = (lane < 16) ? 0 : 8;       // K base per ISA 16-bit layout
  int mofs = (lane < 16) ? 0 : 8;       // C-layout row offset

  v8f acc[2][4] = {};

  for (int kt = 0; kt < K; kt += 32) {
    // ---- stage A tile (256 rows x 32 k) ----
#pragma unroll
    for (int u = 0; u < 2; ++u) {
      int t2 = tid + u * 256;           // 0..511 : (row, 16-wide k-half)
      int r  = t2 >> 1;
      int kh = (t2 & 1) << 4;
      int gr = row0 + r;
      float f[16];
      if (gr < M) {
        const float* ga = &A[(size_t)gr * lda + kt + kh];
#pragma unroll
        for (int j = 0; j < 4; ++j) {
          float4 q = *(const float4*)(ga + 4 * j);
          f[4 * j + 0] = q.x; f[4 * j + 1] = q.y;
          f[4 * j + 2] = q.z; f[4 * j + 3] = q.w;
        }
      } else {
#pragma unroll
        for (int j = 0; j < 16; ++j) f[j] = 0.0f;
      }
      st8(&As[r][kh], f);
      st8(&As[r][kh + 8], f + 8);
    }
    // ---- stage B tile transposed: Bst[c][r], coalesced reads over columns --
    {
      int c  = tid & 63;
      int rb = (tid >> 6) << 3;         // 0,8,16,24
      int gc = col0 + c;
      float f[8];
#pragma unroll
      for (int j = 0; j < 8; ++j)
        f[j] = (gc < N) ? Bm[(size_t)(kt + rb + j) * ldb + gc] : 0.0f;
      st8(&Bst[c][rb], f);
    }
    __syncthreads();

    // ---- prefetch next K tile (global_prefetch_b8) ----
    if (kt + 32 < K) {
      int pr = row0 + tid;
      if (pr < M) __builtin_prefetch(&A[(size_t)pr * lda + kt + 32], 0, 1);
      int pc = col0 + ((tid & 7) << 3);
      if (pc < N) __builtin_prefetch(&Bm[(size_t)(kt + 32 + (tid >> 3)) * ldb + pc], 0, 1);
    }

    // ---- fragments: 12 x ds_load_b128 -> 8 x v_wmma ----
    v16bf bfr[4];
#pragma unroll
    for (int ns = 0; ns < 4; ++ns)
      bfr[ns] = ld_frag(&Bst[ns * 16 + nlo][kb]);
    v16bf afr[2];
#pragma unroll
    for (int mi = 0; mi < 2; ++mi)
      afr[mi] = ld_frag(&As[wave * 32 + mi * 16 + nlo][kb]);
#pragma unroll
    for (int mi = 0; mi < 2; ++mi)
#pragma unroll
      for (int ns = 0; ns < 4; ++ns)
        acc[mi][ns] = __builtin_amdgcn_wmma_f32_16x16x32_bf16(
            false, afr[mi], false, bfr[ns], (short)0, acc[mi][ns], false, false);
    __syncthreads();
  }

  // ---- epilogue: bias / residual / relu, N-edge guarded ----
#pragma unroll
  for (int mi = 0; mi < 2; ++mi)
#pragma unroll
    for (int ns = 0; ns < 4; ++ns)
#pragma unroll
      for (int i = 0; i < 8; ++i) {
        int gm = row0 + wave * 32 + mi * 16 + mofs + i;
        int gn = col0 + ns * 16 + nlo;
        if (gm < M && gn < N) {
          float v = acc[mi][ns][i];
          if (bias)  v += bias[gn];
          if (resid) v += resid[(size_t)gm * ldr + gn];
          if (relu && v < 0.0f) v = 0.0f;
          C[(size_t)gm * ldc + gn] = v;
        }
      }
}

// ---------------------------------------------------------------------------
// Flash attention (causal). Q/K/V stored [NTOK, D] with column = h*HD + hd.
// Block = (q-strip of 128, one (b,h)); 8 waves, each owns 16 query rows.
// K tile row-major (K^T frags contiguous); V tile stored transposed so P@V
// B-fragments are contiguous. All fragments load as 2 x ds_load_b128.
// ---------------------------------------------------------------------------
__global__ __launch_bounds__(256) void attn_k(const float* __restrict__ Q,
                                              const float* __restrict__ Kp,
                                              const float* __restrict__ Vp,
                                              float* __restrict__ att,
                                              float scale) {
  __shared__ alignas(16) __bf16 Kt[32][72];     // [key][hd], stride 144B
  __shared__ alignas(16) __bf16 Vtt[64][40];    // transposed: [hd][key]
  __shared__ alignas(16) __bf16 Pt[8][16][40];  // per-wave P tile 16x32

  int tid  = threadIdx.x;
  int wave = tid >> 5;
  int lane = tid & 31;
  int qblk = blockIdx.x;
  int bh   = blockIdx.y;
  int bat  = bh >> 3;
  int h    = bh & 7;
  int q0   = qblk * 128 + wave * 16;

  int nlo  = lane & 15;
  int kb   = (lane < 16) ? 0 : 8;
  int mofs = (lane < 16) ? 0 : 8;

  // Q as two 16x32 A fragments (hd 0..31, 32..63), held in registers
  v16bf aQ0, aQ1;
  {
    size_t qrow = (size_t)(bat * TT + q0 + nlo) * DD + h * HD;
#pragma unroll
    for (int e = 0; e < 16; ++e) {
      int kk = kb + e + ((e >= 8) ? 8 : 0);
      aQ0[e] = (__bf16)Q[qrow + kk];
      aQ1[e] = (__bf16)Q[qrow + 32 + kk];
    }
  }

  v8f O[4] = {};
  float mrow[8], lrow[8];
#pragma unroll
  for (int i = 0; i < 8; ++i) { mrow[i] = -INFINITY; lrow[i] = 0.0f; }

  int nkb = 4 * (qblk + 1);                      // causal: key blocks needed
  for (int kbi = 0; kbi < nkb; ++kbi) {
    int kbase = kbi * 32;
    // ---- stage K (row-major) ----
    {
      int r   = tid >> 3;                        // 0..31 key
      int hdb = (tid & 7) << 3;                  // 0..56 hd
      const float* gk = &Kp[(size_t)(bat * TT + kbase + r) * DD + h * HD + hdb];
      float f[8];
#pragma unroll
      for (int j = 0; j < 2; ++j) {
        float4 q = *(const float4*)(gk + 4 * j);
        f[4 * j + 0] = q.x; f[4 * j + 1] = q.y;
        f[4 * j + 2] = q.z; f[4 * j + 3] = q.w;
      }
      st8(&Kt[r][hdb], f);
    }
    // ---- stage V transposed: Vtt[hd][key] ----
    {
      int c  = tid & 63;                         // hd
      int rb = (tid >> 6) << 3;                  // key base 0,8,16,24
      float f[8];
#pragma unroll
      for (int j = 0; j < 8; ++j)
        f[j] = Vp[(size_t)(bat * TT + kbase + rb + j) * DD + h * HD + c];
      st8(&Vtt[c][rb], f);
    }
    __syncthreads();

    if (kbase <= q0 + 15) {                      // wave-uniform predicate
      // S = Q K^T : two 16x16 tiles, K-dim 64 = 2 WMMA steps each
      v8f s[2];
#pragma unroll
      for (int ns = 0; ns < 2; ++ns) {
        v16bf bk0 = ld_frag(&Kt[ns * 16 + nlo][kb]);
        v16bf bk1 = ld_frag(&Kt[ns * 16 + nlo][32 + kb]);
        v8f z = {};
        s[ns] = __builtin_amdgcn_wmma_f32_16x16x32_bf16(
            false, aQ0, false, bk0, (short)0, z, false, false);
        s[ns] = __builtin_amdgcn_wmma_f32_16x16x32_bf16(
            false, aQ1, false, bk1, (short)0, s[ns], false, false);
      }

      // scale + causal mask (element i -> row q0+mofs+i, col kbase+ns*16+nlo)
      float sv[2][8];
#pragma unroll
      for (int ns = 0; ns < 2; ++ns) {
        int key = kbase + ns * 16 + nlo;
#pragma unroll
        for (int i = 0; i < 8; ++i) {
          float v = s[ns][i] * scale;
          if (key > q0 + mofs + i) v = -INFINITY;
          sv[ns][i] = v;
        }
      }

      // row max over 32 keys: butterfly across the 16-lane half
      float rm[8];
#pragma unroll
      for (int i = 0; i < 8; ++i) rm[i] = fmaxf(sv[0][i], sv[1][i]);
#pragma unroll
      for (int d = 1; d <= 8; d <<= 1)
#pragma unroll
        for (int i = 0; i < 8; ++i)
          rm[i] = fmaxf(rm[i], __shfl_xor(rm[i], d, 32));

      float alpha[8];
#pragma unroll
      for (int i = 0; i < 8; ++i) {
        float mnew = fmaxf(mrow[i], rm[i]);
        alpha[i] = __expf(mrow[i] - mnew);
        mrow[i] = mnew;
      }

      // P = exp(S - m), row sums
      float p[2][8], rs[8];
#pragma unroll
      for (int i = 0; i < 8; ++i) {
        p[0][i] = __expf(sv[0][i] - mrow[i]);
        p[1][i] = __expf(sv[1][i] - mrow[i]);
        rs[i] = p[0][i] + p[1][i];
      }
#pragma unroll
      for (int d = 1; d <= 8; d <<= 1)
#pragma unroll
        for (int i = 0; i < 8; ++i) rs[i] += __shfl_xor(rs[i], d, 32);
#pragma unroll
      for (int i = 0; i < 8; ++i) lrow[i] = lrow[i] * alpha[i] + rs[i];

      // rescale running O
#pragma unroll
      for (int t = 0; t < 4; ++t)
#pragma unroll
        for (int i = 0; i < 8; ++i) O[t][i] *= alpha[i];

      // P -> per-wave LDS (C-layout store), reload as 16x32 A fragment
#pragma unroll
      for (int ns = 0; ns < 2; ++ns)
#pragma unroll
        for (int i = 0; i < 8; ++i)
          Pt[wave][mofs + i][ns * 16 + nlo] = (__bf16)p[ns][i];

      v16bf pf = ld_frag(&Pt[wave][nlo][kb]);
      // O += P V : 4 output subtiles over hd (B-frags contiguous in Vtt)
#pragma unroll
      for (int t = 0; t < 4; ++t) {
        v16bf vf = ld_frag(&Vtt[t * 16 + nlo][kb]);
        O[t] = __builtin_amdgcn_wmma_f32_16x16x32_bf16(
            false, pf, false, vf, (short)0, O[t], false, false);
      }
    }
    __syncthreads();
  }

  // normalize and store to head-concat layout att[bt, h*HD + hd]
#pragma unroll
  for (int i = 0; i < 8; ++i) {
    float inv = 1.0f / lrow[i];
    size_t row = (size_t)(bat * TT + q0 + mofs + i) * DD + h * HD;
#pragma unroll
    for (int t = 0; t < 4; ++t) att[row + t * 16 + nlo] = O[t][i] * inv;
  }
}

// ---------------------------------------------------------------------------
// Host-side orchestration
// ---------------------------------------------------------------------------
extern "C" void kernel_launch(void* const* d_in, const int* in_sizes, int n_in,
                              void* d_out, int out_size, void* d_ws, size_t ws_size,
                              hipStream_t stream) {
  (void)in_sizes; (void)n_in; (void)out_size; (void)ws_size;

  const int*   idx  = (const int*)  d_in[0];
  const float* tok  = (const float*)d_in[1];
  const float* pos  = (const float*)d_in[2];
  const float* Wq   = (const float*)d_in[3];
  const float* Wk   = (const float*)d_in[4];
  const float* Wv   = (const float*)d_in[5];
  const float* Wo   = (const float*)d_in[6];
  const float* bo   = (const float*)d_in[7];
  const float* W1   = (const float*)d_in[8];
  const float* b1   = (const float*)d_in[9];
  const float* W2   = (const float*)d_in[10];
  const float* b2   = (const float*)d_in[11];
  const float* ln1w = (const float*)d_in[12];
  const float* ln1b = (const float*)d_in[13];
  const float* ln2w = (const float*)d_in[14];
  const float* ln2b = (const float*)d_in[15];
  const float* Wlm  = (const float*)d_in[16];
  const float* blm  = (const float*)d_in[17];
  float* out = (float*)d_out;

  // workspace layout (floats): 11 * NTOK*DD  ~= 44 MB
  const size_t ND = (size_t)NTOK * DD;
  float* base = (float*)d_ws;
  float* x    = base;
  float* hln  = base + 1 * ND;
  float* Qb   = base + 2 * ND;
  float* Kb   = base + 3 * ND;
  float* Vb   = base + 4 * ND;
  float* attb = base + 5 * ND;
  float* h2   = base + 6 * ND;
  float* ff   = base + 7 * ND;   // NTOK x FFD = 4*ND, ends at 11*ND

  const float scale = 1.0f / sqrtf((float)DD);   // D^-0.5 (full n_embd)

  embed_k<<<(NTOK * DD) / 256, 256, 0, stream>>>(idx, tok, pos, x);

  for (int l = 0; l < LL; ++l) {
    layernorm_k<<<NTOK, 256, 0, stream>>>(x, ln1w + l * DD, ln1b + l * DD, hln);

    // QKV projections, per head: [2048,512] @ [512,64] -> head-concat buffers
    for (int h = 0; h < HH; ++h) {
      const size_t wofs = ((size_t)l * HH + h) * DD * HD;
      gemm_bf16_wmma<<<dim3(1, NTOK / 256), 256, 0, stream>>>(
          hln, DD, Wq + wofs, HD, nullptr, nullptr, 0, Qb + h * HD, DD,
          NTOK, HD, DD, 0);
      gemm_bf16_wmma<<<dim3(1, NTOK / 256), 256, 0, stream>>>(
          hln, DD, Wk + wofs, HD, nullptr, nullptr, 0, Kb + h * HD, DD,
          NTOK, HD, DD, 0);
      gemm_bf16_wmma<<<dim3(1, NTOK / 256), 256, 0, stream>>>(
          hln, DD, Wv + wofs, HD, nullptr, nullptr, 0, Vb + h * HD, DD,
          NTOK, HD, DD, 0);
    }

    attn_k<<<dim3(TT / 128, BB * HH), 256, 0, stream>>>(Qb, Kb, Vb, attb, scale);

    // x = hln + att @ Wo + bo
    gemm_bf16_wmma<<<dim3(DD / 64, NTOK / 256), 256, 0, stream>>>(
        attb, DD, Wo + (size_t)l * DD * DD, DD, bo + l * DD, hln, DD, x, DD,
        NTOK, DD, DD, 0);

    layernorm_k<<<NTOK, 256, 0, stream>>>(x, ln2w + l * DD, ln2b + l * DD, h2);

    // ff = relu(h2 @ W1 + b1)
    gemm_bf16_wmma<<<dim3(FFD / 64, NTOK / 256), 256, 0, stream>>>(
        h2, DD, W1 + (size_t)l * DD * FFD, FFD, b1 + l * FFD, nullptr, 0, ff, FFD,
        NTOK, FFD, DD, 1);

    // x = h2 + ff @ W2 + b2
    gemm_bf16_wmma<<<dim3(DD / 64, NTOK / 256), 256, 0, stream>>>(
        ff, FFD, W2 + (size_t)l * FFD * DD, DD, b2 + l * DD, h2, DD, x, DD,
        NTOK, DD, FFD, 0);
  }

  // logits = x @ Wlm + blm  (N-edge handled in kernel)
  gemm_bf16_wmma<<<dim3((VV + 63) / 64, NTOK / 256), 256, 0, stream>>>(
      x, DD, Wlm, VV, blm, nullptr, 0, out, VV,
      NTOK, VV, DD, 0);
}
